// PaiNN_78323023610152
// MI455X (gfx1250) — compile-verified, matrix operand order checked by reading, not run on hardware
//
#include <hip/hip_runtime.h>
#include <math.h>

#define NATOMS 25000
#define NEDGES 200000
#define FDIM   128
#define F3     (3*FDIM)      // 384
#define NRBF   64
#define CUTOFF 5.0f
#define PI_F   3.14159265358979f

typedef __attribute__((ext_vector_type(16))) _Float16 v16h;
typedef __attribute__((ext_vector_type(8)))  float    v8f;

// ---------------- WMMA fragment helpers (ISA 7.12.2 layouts, wave32) ---------

// A: 16x32 f16 tile from LDS, row-major, lda elements.
// Per-lane elements are two contiguous 16B groups -> compiler emits 2x ds_load_b128.
__device__ __forceinline__ v16h frag_a(const _Float16* A, int lda) {
  const int lane = threadIdx.x & 31;
  const int row  = lane & 15;
  const int half = lane >> 4;
  v16h a;
#pragma unroll
  for (int v = 0; v < 8; ++v) {
    const int kb = ((v & 4) << 2) + half * 8 + 2 * (v & 3);
    a[2*v]   = A[row * lda + kb];
    a[2*v+1] = A[row * lda + kb + 1];
  }
  return a;
}

// B: pre-swizzled tile (512 halves): lane-major, 16 contiguous halves per lane.
// One 32B-per-lane vector load -> 2x global_load_b128.
__device__ __forceinline__ v16h frag_bp(const _Float16* __restrict__ Bp) {
  return *(const v16h*)(Bp + (size_t)(threadIdx.x & 31) * 16);
}

__device__ __forceinline__ v8f wmma32(v16h a, v16h b, v8f c) {
  // (neg_a, A, neg_b, B, c_mod, C, reuse_a, reuse_b)
  return __builtin_amdgcn_wmma_f32_16x16x32_f16(false, a, false, b, (short)0, c,
                                                false, false);
}

__device__ __forceinline__ float silu(float x) {
  return x / (1.0f + __expf(-x));
}

// ---------------- weight f32 -> packed-f16 fragment layout -------------------
// src: [K, Nout] row-major f32 (K may stack layers; K % 32 == 0, Nout % 16 == 0)
// dst: tiles of 512 halves: tile (kt, nt) -> [lane 0..31][elem 0..15]
__global__ void pack_b_kernel(const float* __restrict__ src,
                              _Float16* __restrict__ dst,
                              int K, int Nout) {
  int d = blockIdx.x * blockDim.x + threadIdx.x;
  if (d >= K * Nout) return;
  int tile   = d >> 9;
  int within = d & 511;
  int lane = within >> 4;
  int elem = within & 15;
  int v  = elem >> 1, lo = elem & 1;
  int half = lane >> 4, col = lane & 15;
  int kb = ((v & 4) << 2) + half * 8 + 2 * (v & 3) + lo;
  int ntiles = Nout >> 4;
  int kt = tile / ntiles, nt = tile - kt * ntiles;
  dst[d] = (_Float16)src[(size_t)(kt * 32 + kb) * Nout + nt * 16 + col];
}

// ---------------- init: q = emb[z], mu = 0 -----------------------------------
__global__ void init_kernel(const int* __restrict__ z,
                            const float* __restrict__ emb,
                            float* __restrict__ q,
                            float* __restrict__ mu) {
  int idx = blockIdx.x * blockDim.x + threadIdx.x;
  if (idx < NATOMS * F3) mu[idx] = 0.0f;
  if (idx < NATOMS * FDIM) {
    int n = idx >> 7, f = idx & 127;
    q[idx] = emb[z[n] * FDIM + f];
  }
}

// ---------------- K1: edge geometry + RBF + filter GEMM ----------------------
// 1 block = 16 edges. filters[e,c] = ((phi @ W)[e,c] + b[c]) * fcut[e], f16 out.
__global__ void edge_filter_kernel(const float* __restrict__ pos,
                                   const int* __restrict__ eidx,
                                   const _Float16* __restrict__ Wf,  // packed
                                   const float* __restrict__ bf,     // [384]
                                   _Float16* __restrict__ filters,   // [E,384]
                                   float* __restrict__ dirv) {       // [E,3]
  __shared__ _Float16 sphi[16][NRBF];
  __shared__ _Float16 sres[16][F3];
  __shared__ float sd[16];
  __shared__ float sfc[16];
  const int tid   = threadIdx.x;
  const int ebase = blockIdx.x * 16;

  if (tid < 16) {
    const int e = ebase + tid;
    const int i = eidx[e];
    const int j = eidx[NEDGES + e];
    float rx = pos[i*3+0] - pos[j*3+0];
    float ry = pos[i*3+1] - pos[j*3+1];
    float rz = pos[i*3+2] - pos[j*3+2];
    float d  = sqrtf(rx*rx + ry*ry + rz*rz);
    float inv = 1.0f / d;
    dirv[e*3+0] = rx*inv; dirv[e*3+1] = ry*inv; dirv[e*3+2] = rz*inv;
    sd[tid] = d;
    float fc = 0.5f * (__cosf(d * (PI_F / CUTOFF)) + 1.0f);
    sfc[tid] = (d < CUTOFF) ? fc : 0.0f;
  }
  __syncthreads();
  {
    const float delta = CUTOFF / (float)(NRBF - 1);
    const float coeff = -0.5f / (delta * delta);
#pragma unroll
    for (int u = 0; u < 4; ++u) {
      int idx = tid * 4 + u;                  // 1024 = 16*64
      int t = idx >> 6, k = idx & 63;
      float diff = sd[t] - (float)k * delta;
      sphi[t][k] = (_Float16)__expf(coeff * diff * diff);
    }
  }
  __syncthreads();
  const int wave = tid >> 5;
  const int lane = tid & 31;
  const int col  = lane & 15;
  const int half = lane >> 4;
#pragma unroll
  for (int r = 0; r < 3; ++r) {
    const int nt = wave * 3 + r;              // 24 n-tiles of 16 -> 384
    v8f acc = {};
#pragma unroll
    for (int ks = 0; ks < 2; ++ks) {          // K = 64
      v16h a = frag_a(&sphi[0][ks * 32], NRBF);
      v16h b = frag_bp(Wf + (size_t)(ks * 24 + nt) * 512);
      acc = wmma32(a, b, acc);
    }
    const int c = nt * 16 + col;
    const float bias = bf[c];
#pragma unroll
    for (int v = 0; v < 8; ++v) {
      const int m = v + half * 8;
      sres[m][c] = (_Float16)((acc[v] + bias) * sfc[m]);
    }
  }
  __syncthreads();
  {
    const uint4* s4 = (const uint4*)&sres[0][0];
    uint4* g4 = (uint4*)(filters + (size_t)ebase * F3);
#pragma unroll
    for (int u = 0; u < 3; ++u) {             // 768 uint4 = 16*384 halves
      int i4 = tid + u * 256;
      g4[i4] = s4[i4];
    }
  }
}

// ---------------- K2: inter-atomic MLP: x = silu(qW1+b1)W2+b2 ----------------
__global__ void inter_mlp_kernel(const float* __restrict__ q,
                                 const _Float16* __restrict__ W1, // packed
                                 const float* __restrict__ b1,    // [128]
                                 const _Float16* __restrict__ W2, // packed
                                 const float* __restrict__ b2,    // [384]
                                 float* __restrict__ x) {         // [N,384]
  __shared__ _Float16 sq[16][FDIM];
  __shared__ _Float16 sh[16][FDIM];
  __shared__ float    sx2[16][F3];
  const int tid  = threadIdx.x;
  const int base = blockIdx.x * 16;
#pragma unroll
  for (int u = 0; u < 8; ++u) {
    int idx = tid + u * 256;                  // 2048 = 16*128
    int t = idx >> 7, f = idx & 127;
    int row = base + t;
    sq[t][f] = (_Float16)((row < NATOMS) ? q[(size_t)row * FDIM + f] : 0.0f);
  }
  __syncthreads();
  const int wave = tid >> 5;
  const int lane = tid & 31;
  const int col  = lane & 15;
  const int half = lane >> 4;
  {
    const int nt = wave;                      // 8 n-tiles -> 128
    v8f acc = {};
#pragma unroll
    for (int ks = 0; ks < 4; ++ks) {          // K = 128
      v16h a = frag_a(&sq[0][ks * 32], FDIM);
      v16h b = frag_bp(W1 + (size_t)(ks * 8 + nt) * 512);
      acc = wmma32(a, b, acc);
    }
    const int c = nt * 16 + col;
    const float bias = b1[c];
#pragma unroll
    for (int v = 0; v < 8; ++v) {
      const int m = v + half * 8;
      sh[m][c] = (_Float16)silu(acc[v] + bias);
    }
  }
  __syncthreads();
#pragma unroll
  for (int r = 0; r < 3; ++r) {
    const int nt = wave * 3 + r;              // 24 n-tiles -> 384
    v8f acc = {};
#pragma unroll
    for (int ks = 0; ks < 4; ++ks) {
      v16h a = frag_a(&sh[0][ks * 32], FDIM);
      v16h b = frag_bp(W2 + (size_t)(ks * 24 + nt) * 512);
      acc = wmma32(a, b, acc);
    }
    const int c = nt * 16 + col;
    const float bias = b2[c];
#pragma unroll
    for (int v = 0; v < 8; ++v) {
      const int m = v + half * 8;
      sx2[m][c] = acc[v] + bias;
    }
  }
  __syncthreads();
  {
    const float4* s4 = (const float4*)&sx2[0][0];
    float4* g4 = (float4*)(x + (size_t)base * F3);
    const int rmax = NATOMS - base;
#pragma unroll
    for (int u = 0; u < 6; ++u) {             // 1536 float4 = 16 rows * 96
      int i4 = tid + u * 256;
      int r  = i4 / 96;
      if (r < rmax) g4[i4] = s4[i4];
    }
  }
}

// ---------------- K3: edge messages + contiguous segment-sum -----------------
// idx_i = node*8+e (contiguous), so each (node, feature) thread accumulates its
// 8 edges in registers. Reads muIn, writes muOut (pre-update gather semantics).
__global__ void edge_message_kernel(const int* __restrict__ eidx,
                                    const _Float16* __restrict__ filters,
                                    const float* __restrict__ x,
                                    const float* __restrict__ dirv,
                                    const float* __restrict__ muIn,
                                    float* __restrict__ muOut,
                                    float* __restrict__ q) {
  const int tid  = threadIdx.x;
  const int node = blockIdx.x * 2 + (tid >> 7);
  const int f    = tid & 127;
  float accq = 0.0f, a0 = 0.0f, a1 = 0.0f, a2 = 0.0f;
  const int ebase = node * 8;
#pragma unroll
  for (int e = 0; e < 8; ++e) {
    const int eid = ebase + e;
    const int j   = eidx[NEDGES + eid];
    const _Float16* fl = filters + (size_t)eid * F3;
    const float*    xj = x    + (size_t)j * F3;
    const float*    mj = muIn + (size_t)j * F3;
    __builtin_prefetch(fl + F3, 0, 1);        // stream next edge's filter line
    float f1 = (float)fl[f];
    float f2 = (float)fl[FDIM + f];
    float f3 = (float)fl[2*FDIM + f];
    float dq  = f1 * xj[f];
    float dmr = f2 * xj[FDIM + f];
    float dmm = f3 * xj[2*FDIM + f];
    float d0 = dirv[eid*3+0], d1 = dirv[eid*3+1], d2 = dirv[eid*3+2];
    accq += dq;
    a0 += dmr * d0 + dmm * mj[f];
    a1 += dmr * d1 + dmm * mj[FDIM + f];
    a2 += dmr * d2 + dmm * mj[2*FDIM + f];
  }
  q[(size_t)node * FDIM + f] += accq;
  const size_t mb = (size_t)node * F3;
  muOut[mb + f]          = muIn[mb + f]          + a0;
  muOut[mb + FDIM + f]   = muIn[mb + FDIM + f]   + a1;
  muOut[mb + 2*FDIM + f] = muIn[mb + 2*FDIM + f] + a2;
}

// ---------------- K4: mix GEMM [3N,128]x[128,256] + vector norm --------------
__global__ void mix_kernel(const float* __restrict__ mu,     // [3N,128]
                           const _Float16* __restrict__ Wm,  // packed
                           const float* __restrict__ bm,     // [256]
                           float* __restrict__ muV,          // [3N,128]
                           float* __restrict__ muW,          // [3N,128]
                           float* __restrict__ muVn) {       // [N,128]
  __shared__ _Float16 smu[48][FDIM];
  __shared__ float    sv[48][FDIM];
  __shared__ float    sw[48][FDIM];
  const int tid   = threadIdx.x;
  const int base  = blockIdx.x * 16;          // atoms
  const int rbase = base * 3;                 // mu rows
#pragma unroll
  for (int u = 0; u < 24; ++u) {
    int idx = tid + u * 256;                  // 6144 = 48*128
    int t = idx >> 7, f = idx & 127;
    int R = rbase + t;
    smu[t][f] = (_Float16)((R < 3 * NATOMS) ? mu[(size_t)R * FDIM + f] : 0.0f);
  }
  __syncthreads();
  const int wave = tid >> 5;
  const int lane = tid & 31;
  const int col  = lane & 15;
  const int half = lane >> 4;
#pragma unroll
  for (int jj = 0; jj < 6; ++jj) {
    const int job = wave * 6 + jj;            // 48 jobs = 3 m-tiles x 16 n-tiles
    const int mt  = job >> 4;
    const int nt  = job & 15;
    v8f acc = {};
#pragma unroll
    for (int ks = 0; ks < 4; ++ks) {          // K = 128
      v16h a = frag_a(&smu[mt * 16][ks * 32], FDIM);
      v16h b = frag_bp(Wm + (size_t)(ks * 16 + nt) * 512);
      acc = wmma32(a, b, acc);
    }
    const int c = nt * 16 + col;
    const float bias = bm[c];
#pragma unroll
    for (int v = 0; v < 8; ++v) {
      const int m = mt * 16 + v + half * 8;
      float val = acc[v] + bias;
      if (c < FDIM) sv[m][c]        = val;
      else          sw[m][c - FDIM] = val;
    }
  }
  __syncthreads();
  {
    const float4* v4 = (const float4*)&sv[0][0];
    const float4* w4 = (const float4*)&sw[0][0];
    float4* gV = (float4*)(muV + (size_t)rbase * FDIM);
    float4* gW = (float4*)(muW + (size_t)rbase * FDIM);
    const int rmax = 3 * NATOMS - rbase;
#pragma unroll
    for (int u = 0; u < 6; ++u) {             // 1536 float4 = 48 rows * 32
      int i4 = tid + u * 256;
      int r  = i4 >> 5;
      if (r < rmax) { gV[i4] = v4[i4]; gW[i4] = w4[i4]; }
    }
  }
#pragma unroll
  for (int u = 0; u < 8; ++u) {
    int idx = tid + u * 256;                  // 2048 = 16*128
    int a = idx >> 7, f = idx & 127;
    int atom = base + a;
    if (atom < NATOMS) {
      float v0 = sv[a*3+0][f], v1 = sv[a*3+1][f], v2 = sv[a*3+2][f];
      muVn[(size_t)atom * FDIM + f] = sqrtf(v0*v0 + v1*v1 + v2*v2);
    }
  }
}

// ---------------- K5: intra MLP + coupled q/mu update ------------------------
__global__ void intra_kernel(float* __restrict__ q,
                             const float* __restrict__ muVn,
                             const _Float16* __restrict__ W1, // packed [256,128]
                             const float* __restrict__ b1,    // [128]
                             const _Float16* __restrict__ W2, // packed [128,384]
                             const float* __restrict__ b2,    // [384]
                             const float* __restrict__ muV,
                             const float* __restrict__ muW,
                             float* __restrict__ mu) {        // [3N,128] in-place
  __shared__ _Float16 sctx[16][2*FDIM];
  __shared__ _Float16 shid[16][FDIM];
  __shared__ float    sx[16][F3];
  const int tid  = threadIdx.x;
  const int base = blockIdx.x * 16;
#pragma unroll
  for (int u = 0; u < 16; ++u) {
    int idx = tid + u * 256;                  // 4096 = 16*256
    int t = idx >> 8, c = idx & 255;
    int row = base + t;
    float val = 0.0f;
    if (row < NATOMS)
      val = (c < FDIM) ? q[(size_t)row * FDIM + c]
                       : muVn[(size_t)row * FDIM + (c - FDIM)];
    sctx[t][c] = (_Float16)val;
  }
  __syncthreads();
  const int wave = tid >> 5;
  const int lane = tid & 31;
  const int col  = lane & 15;
  const int half = lane >> 4;
  {
    const int nt = wave;                      // 8 n-tiles -> 128
    v8f acc = {};
#pragma unroll
    for (int ks = 0; ks < 8; ++ks) {          // K = 256
      v16h a = frag_a(&sctx[0][ks * 32], 2*FDIM);
      v16h b = frag_bp(W1 + (size_t)(ks * 8 + nt) * 512);
      acc = wmma32(a, b, acc);
    }
    const int c = nt * 16 + col;
    const float bias = b1[c];
#pragma unroll
    for (int v = 0; v < 8; ++v) {
      const int m = v + half * 8;
      shid[m][c] = (_Float16)silu(acc[v] + bias);
    }
  }
  __syncthreads();
#pragma unroll
  for (int r = 0; r < 3; ++r) {
    const int nt = wave * 3 + r;              // 24 n-tiles -> 384
    v8f acc = {};
#pragma unroll
    for (int ks = 0; ks < 4; ++ks) {          // K = 128
      v16h a = frag_a(&shid[0][ks * 32], FDIM);
      v16h b = frag_bp(W2 + (size_t)(ks * 24 + nt) * 512);
      acc = wmma32(a, b, acc);
    }
    const int c = nt * 16 + col;
    const float bias = b2[c];
#pragma unroll
    for (int v = 0; v < 8; ++v) {
      const int m = v + half * 8;
      sx[m][c] = acc[v] + bias;
    }
  }
  __syncthreads();
#pragma unroll
  for (int u = 0; u < 8; ++u) {
    int idx = tid + u * 256;                  // 2048 = 16*128
    int a = idx >> 7, f = idx & 127;
    int atom = base + a;
    if (atom >= NATOMS) continue;
    float dq   = sx[a][f];
    float dmuc = sx[a][FDIM + f];
    float dqmu = sx[a][2*FDIM + f];
    size_t R = (size_t)atom * 3;
    float s = 0.0f;
#pragma unroll
    for (int d = 0; d < 3; ++d)
      s += muV[(R + d) * FDIM + f] * muW[(R + d) * FDIM + f];
    q[(size_t)atom * FDIM + f] += dq + dqmu * s;
#pragma unroll
    for (int d = 0; d < 3; ++d)
      mu[(R + d) * FDIM + f] += dmuc * muW[(R + d) * FDIM + f];
  }
}

// ---------------- host launcher ----------------------------------------------
extern "C" void kernel_launch(void* const* d_in, const int* in_sizes, int n_in,
                              void* d_out, int out_size, void* d_ws, size_t ws_size,
                              hipStream_t stream) {
  (void)in_sizes; (void)n_in; (void)out_size; (void)ws_size;
  const int*   z      = (const int*)  d_in[0];
  const float* pos    = (const float*)d_in[1];
  const int*   eidx   = (const int*)  d_in[3];
  const float* emb    = (const float*)d_in[4];
  const float* filt_W = (const float*)d_in[5];
  const float* filt_b = (const float*)d_in[6];
  const float* iW1    = (const float*)d_in[7];
  const float* ib1    = (const float*)d_in[8];
  const float* iW2    = (const float*)d_in[9];
  const float* ib2    = (const float*)d_in[10];
  const float* mW     = (const float*)d_in[11];
  const float* mb     = (const float*)d_in[12];
  const float* aW1    = (const float*)d_in[13];
  const float* ab1    = (const float*)d_in[14];
  const float* aW2    = (const float*)d_in[15];
  const float* ab2    = (const float*)d_in[16];

  float* q = (float*)d_out;                   // q lives directly in d_out

  char* p = (char*)d_ws;
  auto take = [&](size_t bytes) -> char* {
    char* r = p; p += (bytes + 255) & ~(size_t)255; return r;
  };
  _Float16* wfilt   = (_Float16*)take((size_t)64*384*2);
  _Float16* wi1     = (_Float16*)take((size_t)3*128*128*2);
  _Float16* wi2     = (_Float16*)take((size_t)3*128*384*2);
  _Float16* wm      = (_Float16*)take((size_t)3*128*256*2);
  _Float16* wa1     = (_Float16*)take((size_t)3*256*128*2);
  _Float16* wa2     = (_Float16*)take((size_t)3*128*384*2);
  _Float16* filters = (_Float16*)take((size_t)NEDGES*F3*2);
  float*    dirv    = (float*)   take((size_t)NEDGES*3*4);
  float*    xbuf    = (float*)   take((size_t)NATOMS*F3*4);
  float*    muA     = (float*)   take((size_t)NATOMS*F3*4);
  float*    muB     = (float*)   take((size_t)NATOMS*F3*4);
  float*    muV     = (float*)   take((size_t)NATOMS*F3*4);
  float*    muWc    = (float*)   take((size_t)NATOMS*F3*4);
  float*    muVn    = (float*)   take((size_t)NATOMS*FDIM*4);

  auto pack = [&](const float* s, _Float16* d, int K, int Nout) {
    int n = K * Nout;
    pack_b_kernel<<<(n + 255) / 256, 256, 0, stream>>>(s, d, K, Nout);
  };
  // stacked layers pack as K-stacked matrices (K%32==0 so tiles never cross a
  // layer; packed layer offset == original l*K*Nout)
  pack(filt_W, wfilt,  64, 384);
  pack(iW1,    wi1,   384, 128);
  pack(iW2,    wi2,   384, 384);
  pack(mW,     wm,    384, 256);
  pack(aW1,    wa1,   768, 128);
  pack(aW2,    wa2,   384, 384);

  init_kernel<<<(NATOMS*F3 + 255) / 256, 256, 0, stream>>>(z, emb, q, muA);

  edge_filter_kernel<<<NEDGES / 16, 256, 0, stream>>>(pos, eidx, wfilt, filt_b,
                                                      filters, dirv);

  const int nb16 = (NATOMS + 15) / 16;
  float* muIn = muA;
  float* muOut = muB;
  for (int l = 0; l < 3; ++l) {
    inter_mlp_kernel<<<nb16, 256, 0, stream>>>(
        q, wi1 + (size_t)l*128*128, ib1 + l*128,
        wi2 + (size_t)l*128*384, ib2 + l*384, xbuf);
    edge_message_kernel<<<NATOMS / 2, 256, 0, stream>>>(
        eidx, filters, xbuf, dirv, muIn, muOut, q);
    mix_kernel<<<nb16, 256, 0, stream>>>(
        muOut, wm + (size_t)l*128*256, mb + l*256, muV, muWc, muVn);
    intra_kernel<<<nb16, 256, 0, stream>>>(
        q, muVn, wa1 + (size_t)l*256*128, ab1 + l*128,
        wa2 + (size_t)l*128*384, ab2 + l*384, muV, muWc, muOut);
    float* t = muIn; muIn = muOut; muOut = t;   // ping-pong mu buffers
  }
}
